// DelayLMLIFSNN_56272661512527
// MI455X (gfx1250) — matrix operand order repaired and checked by reading, not compiled, not run on hardware
//
#include <hip/hip_runtime.h>
#include <hip/hip_bf16.h>

// Model dims
#define T_DIM 512
#define B_DIM 64
#define J_DIM 256
#define H_DIM 256
#define K_DIM 25
#define N_LAYERS 3
#define TT 4            // time steps per conv workgroup
#define NBOUND 6        // first NBOUND time-blocks need causal boundary checks

typedef __attribute__((ext_vector_type(16))) _Float16 v16h;
typedef __attribute__((ext_vector_type(8)))  _Float16 v8h;
typedef __attribute__((ext_vector_type(8)))  float    v8f;

union Frag16 { v16h f; v8h h[2]; };

// ---------------------------------------------------------------------------
// Kernel 0: convert f32 binary spikes -> f16
// ---------------------------------------------------------------------------
__global__ void cvt_f32_to_f16(const float* __restrict__ in,
                               _Float16* __restrict__ out, int n) {
    int i = blockIdx.x * blockDim.x + threadIdx.x;
    if (i < n) out[i] = (_Float16)in[i];
}

// ---------------------------------------------------------------------------
// Kernel 1: build per-shift weight matrices Kd[d][o][j] (f16)
// Kd[d][o][j] = W[o][j] * gauss[o][j][K-1-d] / (sum_k gauss + 1e-7)
// ---------------------------------------------------------------------------
__global__ void build_kd(const float* __restrict__ W, const float* __restrict__ P,
                         _Float16* __restrict__ Kd) {
    int idx = blockIdx.x * blockDim.x + threadIdx.x;   // o*J + j
    if (idx >= H_DIM * J_DIM) return;
    float w = W[idx];
    float c = P[idx] + (float)(K_DIM / 2);
    const float s = 0.5f + 0.27f;                      // |sigma| + 0.27
    float g[K_DIM];
    float sum = 0.f;
    #pragma unroll
    for (int k = 0; k < K_DIM; ++k) {
        float z = ((float)k - c) / s;
        g[k] = __expf(-0.5f * z * z);
        sum += g[k];
    }
    float inv = w / (sum + 1e-7f);
    int o = idx >> 8, j = idx & 255;
    #pragma unroll
    for (int k = 0; k < K_DIM; ++k) {
        int d = K_DIM - 1 - k;                         // shift index
        Kd[((size_t)d * H_DIM + o) * J_DIM + j] = (_Float16)(g[k] * inv);
    }
}

// ---------------------------------------------------------------------------
// WMMA fragment loaders (wave32, 16x16x32 f16 layouts per CDNA5 ISA 7.12.2)
// ---------------------------------------------------------------------------
// A: 16x32 (M=batch rows, K=j). lanes 0-15/16-31 both hold M=0..15;
// halfs 0..7 -> K = hid*8+0..7 ; halfs 8..15 -> K = 16+hid*8+0..7
__device__ __forceinline__ v16h load_a_frag(const _Float16* __restrict__ xrow,
                                            int j0, int lane) {
    int m = lane & 15, hid = lane >> 4;
    const _Float16* p = xrow + (size_t)m * J_DIM + j0 + hid * 8;
    Frag16 f;
    f.h[0] = *(const v8h*)(p);
    f.h[1] = *(const v8h*)(p + 16);
    return f.f;
}

// B: 32x16 (K=j rows, N=o cols). lane holds N = lane&15; halfs 0..7 -> K = hid*8+0..7,
// halfs 8..15 -> K = 16+hid*8+0..7; Kd stored [o][j].
__device__ __forceinline__ v16h load_b_frag(const _Float16* __restrict__ kdd,
                                            int obase, int j0, int lane) {
    int n = lane & 15, hid = lane >> 4;
    const _Float16* p = kdd + (size_t)(obase + n) * J_DIM + j0 + hid * 8;
    Frag16 f;
    f.h[0] = *(const v8h*)(p);
    f.h[1] = *(const v8h*)(p + 16);
    return f.f;
}

// ---------------------------------------------------------------------------
// Kernel 2: delay conv as WMMA GEMM, tiled over TT=4 time steps so the
// time-invariant weight (B) fragments are loaded once and reused 4x.
// grid = (nTimeBlocks, H/64); block = 256 threads = 8 waves.
// wave w: m-tile = w&3 (b0 = 16*(w&3)); n-pair = w>>2 -> o tiles (o0, o0+16).
// Y[t,b,o] = sum_{d,j} X[t-d][b][j] * Kd[d][o][j]
// CHECK=false: tg >= K_DIM-1 guaranteed -> fully branch-free inner loop so
// the scheduler can clause all 12 loads ahead of the 8 WMMAs per j-chunk.
// ---------------------------------------------------------------------------
template<bool CHECK>
__global__ void conv_gemm(const _Float16* __restrict__ X,
                          const _Float16* __restrict__ Kd,
                          float* __restrict__ Y, int tgBase) {
    const int tg   = tgBase + blockIdx.x * TT;
    const int og   = blockIdx.y * 64;
    const int wave = threadIdx.x >> 5;
    const int lane = threadIdx.x & 31;
    const int b0   = (wave & 3) * 16;
    const int o0   = og + (wave >> 2) * 32;   // two adjacent 16-wide n tiles
    const size_t BJ = (size_t)B_DIM * J_DIM;

    v8f acc[TT][2];
    #pragma unroll
    for (int i = 0; i < TT; ++i) { acc[i][0] = (v8f){}; acc[i][1] = (v8f){}; }

    for (int d = 0; d < K_DIM; ++d) {
        if (CHECK && (tg + (TT - 1) - d < 0)) break;   // whole tile out of range
        const _Float16* kdd = Kd + (size_t)d * H_DIM * J_DIM;

        // Prefetch next shift's X rows (gfx1250 global_prefetch_b8 path)
        {
            int tsn = tg + (TT - 1) - (d + 1);
            if (d + 1 < K_DIM && (!CHECK || tsn >= 0)) {
                const _Float16* nx = X + ((size_t)tsn * B_DIM + b0) * J_DIM;
                __builtin_prefetch(nx + lane * 64, 0, 0);
                __builtin_prefetch(nx + 2048 + lane * 64, 0, 0);
            }
        }

        if (!CHECK) {
            // tg - d >= 0 for all d: branch-free hot path
            const _Float16* x0 = X + ((size_t)(tg - d) * B_DIM + b0) * J_DIM;
            for (int j0 = 0; j0 < J_DIM; j0 += 32) {
                v16h bA = load_b_frag(kdd, o0,      j0, lane);
                v16h bB = load_b_frag(kdd, o0 + 16, j0, lane);
                v16h a0 = load_a_frag(x0,          j0, lane);
                v16h a1 = load_a_frag(x0 + BJ,     j0, lane);
                v16h a2 = load_a_frag(x0 + 2 * BJ, j0, lane);
                v16h a3 = load_a_frag(x0 + 3 * BJ, j0, lane);
                acc[0][0] = __builtin_amdgcn_wmma_f32_16x16x32_f16(
                                false, a0, false, bA, (short)0, acc[0][0], false, false);
                acc[0][1] = __builtin_amdgcn_wmma_f32_16x16x32_f16(
                                false, a0, false, bB, (short)0, acc[0][1], false, false);
                acc[1][0] = __builtin_amdgcn_wmma_f32_16x16x32_f16(
                                false, a1, false, bA, (short)0, acc[1][0], false, false);
                acc[1][1] = __builtin_amdgcn_wmma_f32_16x16x32_f16(
                                false, a1, false, bB, (short)0, acc[1][1], false, false);
                acc[2][0] = __builtin_amdgcn_wmma_f32_16x16x32_f16(
                                false, a2, false, bA, (short)0, acc[2][0], false, false);
                acc[2][1] = __builtin_amdgcn_wmma_f32_16x16x32_f16(
                                false, a2, false, bB, (short)0, acc[2][1], false, false);
                acc[3][0] = __builtin_amdgcn_wmma_f32_16x16x32_f16(
                                false, a3, false, bA, (short)0, acc[3][0], false, false);
                acc[3][1] = __builtin_amdgcn_wmma_f32_16x16x32_f16(
                                false, a3, false, bB, (short)0, acc[3][1], false, false);
            }
        } else {
            for (int j0 = 0; j0 < J_DIM; j0 += 32) {
                v16h bA = load_b_frag(kdd, o0,      j0, lane);
                v16h bB = load_b_frag(kdd, o0 + 16, j0, lane);
                #pragma unroll
                for (int i = 0; i < TT; ++i) {
                    int ts = tg + i - d;
                    if (ts < 0) continue;     // wave-uniform: EXEC untouched
                    const _Float16* xrow = X + ((size_t)ts * B_DIM + b0) * J_DIM;
                    v16h a = load_a_frag(xrow, j0, lane);
                    acc[i][0] = __builtin_amdgcn_wmma_f32_16x16x32_f16(
                                    false, a, false, bA, (short)0, acc[i][0], false, false);
                    acc[i][1] = __builtin_amdgcn_wmma_f32_16x16x32_f16(
                                    false, a, false, bB, (short)0, acc[i][1], false, false);
                }
            }
        }
    }

    // C/D layout: VGPR r: lanes 0-15 -> (M=r, N=lane); lanes 16-31 -> (M=8+r)
    const int n     = lane & 15;
    const int mbase = (lane >> 4) * 8;
    #pragma unroll
    for (int i = 0; i < TT; ++i) {
        float* yrow = Y + ((size_t)(tg + i) * B_DIM + b0 + mbase) * H_DIM;
        #pragma unroll
        for (int r = 0; r < 8; ++r) {
            yrow[(size_t)r * H_DIM + o0 + n]      = acc[i][0][r];
            yrow[(size_t)r * H_DIM + o0 + 16 + n] = acc[i][1][r];
        }
    }
}

// ---------------------------------------------------------------------------
// Kernel 3a: BN partial sums. 256 blocks, each reduces 128 rows x 256 channels
// (coalesced across threads = channels). part[0..65535]=sum, [65536..]=sumsq.
// ---------------------------------------------------------------------------
__global__ void bn_part(const float* __restrict__ Y, float* __restrict__ part) {
    const int row0 = blockIdx.x * 128;
    const int c = threadIdx.x;
    float s = 0.f, s2 = 0.f;
    for (int r = 0; r < 128; ++r) {
        float v = Y[(size_t)(row0 + r) * H_DIM + c];
        s += v; s2 += v * v;
    }
    part[blockIdx.x * H_DIM + c]               = s;
    part[256 * H_DIM + blockIdx.x * H_DIM + c] = s2;
}

// Kernel 3b: finalize per-channel scale/shift. 1 block, 256 threads.
__global__ void bn_final(const float* __restrict__ part,
                         const float* __restrict__ gamma,
                         const float* __restrict__ bb,
                         float* __restrict__ scale, float* __restrict__ shift) {
    const int c = threadIdx.x;
    float s = 0.f, s2 = 0.f;
    for (int i = 0; i < 256; ++i) {
        s  += part[i * H_DIM + c];
        s2 += part[256 * H_DIM + i * H_DIM + c];
    }
    const float invN = 1.0f / (float)(T_DIM * B_DIM);
    float mu  = s * invN;
    float var = s2 * invN - mu * mu;
    float sc  = gamma[c] * rsqrtf(var + 1e-5f);
    scale[c] = sc;
    shift[c] = bb[c] - mu * sc;
}

// ---------------------------------------------------------------------------
// Kernel 4: LIF soft-reset scan over time. One thread per (b,h) state.
// U = beta*(U - S) + (1-beta)*yn ; S = (U - 1 > 0)
// ---------------------------------------------------------------------------
__global__ void lif_scan(const float* __restrict__ Y,
                         const float* __restrict__ scale,
                         const float* __restrict__ shift,
                         const float* __restrict__ beta,
                         const float* __restrict__ U0,
                         _Float16* __restrict__ Xnext,
                         float* __restrict__ out, int writeOut) {
    const int idx = blockIdx.x * blockDim.x + threadIdx.x;   // b*H + h
    const int h = idx & (H_DIM - 1);
    const float sc = scale[h], sh = shift[h], be = beta[h];
    float U = U0[idx];
    float S = 0.f;
    for (int t = 0; t < T_DIM; ++t) {
        float yn = Y[(size_t)t * (B_DIM * H_DIM) + idx] * sc + sh;
        U = be * (U - S) + (1.f - be) * yn;
        S = (U - 1.0f > 0.f) ? 1.f : 0.f;
        Xnext[(size_t)t * (B_DIM * H_DIM) + idx] = (_Float16)S;
        if (writeOut) out[(size_t)t * (B_DIM * H_DIM) + idx] = S;
    }
}

// ---------------------------------------------------------------------------
extern "C" void kernel_launch(void* const* d_in, const int* in_sizes, int n_in,
                              void* d_out, int out_size, void* d_ws, size_t ws_size,
                              hipStream_t stream) {
    const float* x     = (const float*)d_in[0];  // (T,B,J)
    const float* W     = (const float*)d_in[1];  // (L,H,J)
    const float* P     = (const float*)d_in[2];  // (L,H,J)
    const float* beta  = (const float*)d_in[3];  // (L,H)
    const float* gamma = (const float*)d_in[4];  // (L,H)
    const float* bb    = (const float*)d_in[5];  // (L,H)
    const float* U0    = (const float*)d_in[6];  // (L,B,H)
    float* out = (float*)d_out;

    char* ws = (char*)d_ws;
    size_t off = 0;
    auto carve = [&](size_t bytes) -> void* {
        void* p = ws + off;
        off += (bytes + 255) & ~(size_t)255;
        return p;
    };
    const size_t TBJ = (size_t)T_DIM * B_DIM * J_DIM;
    _Float16* Xa    = (_Float16*)carve(TBJ * sizeof(_Float16));
    _Float16* Xb    = (_Float16*)carve(TBJ * sizeof(_Float16));
    float*    Y     = (float*)   carve(TBJ * sizeof(float));
    _Float16* Kd    = (_Float16*)carve((size_t)K_DIM * H_DIM * J_DIM * sizeof(_Float16));
    float*    part  = (float*)   carve((size_t)2 * 256 * H_DIM * sizeof(float));
    float*    scale = (float*)   carve(H_DIM * sizeof(float));
    float*    shift = (float*)   carve(H_DIM * sizeof(float));

    // input spikes f32 -> f16
    cvt_f32_to_f16<<<(int)((TBJ + 255) / 256), 256, 0, stream>>>(x, Xa, (int)TBJ);

    _Float16* Xin  = Xa;
    _Float16* Xnxt = Xb;
    for (int l = 0; l < N_LAYERS; ++l) {
        const size_t wOff = (size_t)l * H_DIM * J_DIM;
        build_kd<<<(H_DIM * J_DIM) / 256, 256, 0, stream>>>(W + wOff, P + wOff, Kd);

        // boundary blocks (tg < K-1): checked variant
        conv_gemm<true><<<dim3(NBOUND, H_DIM / 64), 256, 0, stream>>>(
            Xin, Kd, Y, 0);
        // steady-state blocks: branch-free variant
        conv_gemm<false><<<dim3(T_DIM / TT - NBOUND, H_DIM / 64), 256, 0, stream>>>(
            Xin, Kd, Y, NBOUND * TT);

        bn_part<<<256, 256, 0, stream>>>(Y, part);
        bn_final<<<1, 256, 0, stream>>>(part, gamma + l * H_DIM, bb + l * H_DIM,
                                        scale, shift);

        lif_scan<<<(B_DIM * H_DIM) / 256, 256, 0, stream>>>(
            Y, scale, shift, beta + l * H_DIM, U0 + (size_t)l * B_DIM * H_DIM,
            Xnxt, out, (l == N_LAYERS - 1) ? 1 : 0);

        _Float16* tmp = Xin; Xin = Xnxt; Xnxt = tmp;
    }
}
